// MixtureOfExpertsLayer_48902497632839
// MI455X (gfx1250) — matrix operand order repaired and checked by reading, not compiled, not run on hardware
//
#include <hip/hip_runtime.h>
#include <hip/hip_bf16.h>
#include <cstdint>

// ---------------------------------------------------------------------------
// MoE (top-2 of 8 experts) for MI455X / gfx1250, wave32, bf16 WMMA.
//   N=8192 tokens, D=1024, F=2752, E=8, TOP_K=2
// Routed pipeline: gate -> prefix -> slotfix -> per-expert
//   { transpose/cvt weights -> SwiGLU GEMM -> down GEMM } -> combine -> aux.
// GEMMs: v_wmma_f32_16x16x32_bf16, double-buffered LDS tiles fed by
// TENSOR_LOAD_TO_LDS (weights, TENSORcnt) + global_load_async_to_lds_b128
// (gathered activations, ASYNCcnt).
// ---------------------------------------------------------------------------

#define NTOK 8192
#define DDIM 1024
#define NEXP 8
#define FDIM 2752

typedef __attribute__((ext_vector_type(16))) __bf16 v16bf;
typedef __attribute__((ext_vector_type(8)))  float  v8f;
typedef __attribute__((ext_vector_type(4)))  unsigned int v4u;
typedef __attribute__((ext_vector_type(8)))  int v8i;
typedef __attribute__((ext_vector_type(4)))  int v4i;

typedef __attribute__((address_space(1))) v4i as1_v4i;   // global (__device__)
typedef __attribute__((address_space(3))) v4i as3_v4i;   // LDS (__shared__)

static_assert(sizeof(v16bf) == 32, "v16bf must be 32 bytes");

union FragU {
  uint4 u4[2];
  v16bf v;
};

#if __has_builtin(__builtin_amdgcn_global_load_async_to_lds_b128)
#define HAVE_ASYNC 1
#else
#define HAVE_ASYNC 0
#endif

#if __has_builtin(__builtin_amdgcn_tensor_load_to_lds)
#define HAVE_TDM 1
#else
#define HAVE_TDM 0
#endif

#if __has_builtin(__builtin_amdgcn_s_wait_asynccnt)
#define WAIT_ASYNC(n) __builtin_amdgcn_s_wait_asynccnt(n)
#else
#define WAIT_ASYNC(n) asm volatile("s_wait_asynccnt %0" ::"n"(n) : "memory")
#endif

#if __has_builtin(__builtin_amdgcn_s_wait_tensorcnt)
#define WAIT_TENSOR(n) __builtin_amdgcn_s_wait_tensorcnt(n)
#else
#define WAIT_TENSOR(n) asm volatile("s_wait_tensorcnt %0" ::"n"(n) : "memory")
#endif

#if HAVE_TDM
#define K1_ASY 1   // asyncs issued per tile per thread (A only)
#define K2_ASY 1
#else
#define K1_ASY 5   // A(1) + Bg(2) + Bu(2)
#define K2_ASY 3   // A(1) + Bd(2)
#endif

__device__ __forceinline__ unsigned lds_off32(const void* p) {
  return (unsigned)(uintptr_t)(__attribute__((address_space(3))) const void*)p;
}

// 16-byte global -> LDS transfer (async engine when available)
__device__ __forceinline__ void copy16(const void* g, void* l) {
#if HAVE_ASYNC
  __builtin_amdgcn_global_load_async_to_lds_b128(
      (as1_v4i*)(uintptr_t)g, (as3_v4i*)l, 0, 0);
#else
  *(uint4*)l = *(const uint4*)g;
#endif
}

#if HAVE_TDM
// One 2D TDM transfer: tile_rows x 32 bf16 elems, row stride in elements,
// into contiguous LDS at lds_off.  D# packed per CDNA5 ISA 8.3/8.4.
__device__ __forceinline__ void tdm_load_2d(const void* gsrc, unsigned lds_off,
                                            unsigned tile_rows,
                                            unsigned row_stride_elems) {
  unsigned long long g = (unsigned long long)(uintptr_t)gsrc;
  v4u g0;
  g0[0] = 1u;                                              // count=1, user mode
  g0[1] = lds_off;                                         // lds_addr
  g0[2] = (unsigned)g;                                     // global_addr lo
  g0[3] = (unsigned)((g >> 32) & 0x01FFFFFFu) | (2u << 30);// addr hi | type=2
  v8i g1;
  g1[0] = (int)(1u << 16);                 // wg_mask=0, data_size=1 (2 bytes)
  g1[1] = (int)(32u << 16);                // tensor_dim0 = 32 (lo16 in [31:16])
  g1[2] = (int)(tile_rows << 16);          // dim0 hi=0 | tensor_dim1 lo16
  g1[3] = (int)(32u << 16);                // tensor_dim1 hi=0 | tile_dim0=32
  g1[4] = (int)tile_rows;                  // tile_dim1 | tile_dim2=0
  g1[5] = (int)row_stride_elems;           // tensor_dim0_stride lo32
  g1[6] = 0;                               // stride0 hi | stride1 lo
  g1[7] = 0;                               // stride1 hi
  v4i z4 = {0, 0, 0, 0};
  v8i z8 = {0, 0, 0, 0, 0, 0, 0, 0};
  __builtin_amdgcn_tensor_load_to_lds(g0, g1, z4, z4, z8, 0);
}
#endif

__device__ __forceinline__ unsigned short f2bf(float f) {
  unsigned u = __float_as_uint(f);
  unsigned r = u + 0x7fffu + ((u >> 16) & 1u);  // round-to-nearest-even
  return (unsigned short)(r >> 16);
}

// ---------------------------------------------------------------------------
// Gate: logits = x @ gate_w, softmax, top-2, renorm; x -> bf16; routing slot
// assignment (int atomics); aux-loss partial sums. One wave per token.
// ---------------------------------------------------------------------------
__global__ __launch_bounds__(256) void gate_kernel(
    const float* __restrict__ x, const float* __restrict__ gw,
    unsigned short* __restrict__ xb,
    float* __restrict__ wtk, int* __restrict__ eid, int* __restrict__ lpos,
    int* __restrict__ cnt, float* __restrict__ sumP) {
  __shared__ float sP[NEXP];
  const int tid = threadIdx.x;
  if (tid < NEXP) sP[tid] = 0.0f;
  __syncthreads();

  const int wave = tid >> 5, lane = tid & 31;
  const int t = blockIdx.x * 8 + wave;

  float acc[NEXP];
#pragma unroll
  for (int e = 0; e < NEXP; ++e) acc[e] = 0.0f;

  for (int d = lane; d < DDIM; d += 32) {
    float xv = x[(size_t)t * DDIM + d];
    xb[(size_t)t * DDIM + d] = f2bf(xv);
    const float4* g4 = (const float4*)(gw + (size_t)d * NEXP);
    float4 a = g4[0], b = g4[1];
    acc[0] += xv * a.x; acc[1] += xv * a.y; acc[2] += xv * a.z; acc[3] += xv * a.w;
    acc[4] += xv * b.x; acc[5] += xv * b.y; acc[6] += xv * b.z; acc[7] += xv * b.w;
  }
#pragma unroll
  for (int e = 0; e < NEXP; ++e) {
#pragma unroll
    for (int off = 16; off > 0; off >>= 1) acc[e] += __shfl_xor(acc[e], off, 32);
  }

  if (lane == 0) {
    float mx = acc[0];
#pragma unroll
    for (int e = 1; e < NEXP; ++e) mx = fmaxf(mx, acc[e]);
    float p[NEXP], s = 0.0f;
#pragma unroll
    for (int e = 0; e < NEXP; ++e) { p[e] = __expf(acc[e] - mx); s += p[e]; }
    float inv = 1.0f / s;
#pragma unroll
    for (int e = 0; e < NEXP; ++e) p[e] *= inv;

    int e0 = 0;
#pragma unroll
    for (int e = 1; e < NEXP; ++e) if (p[e] > p[e0]) e0 = e;
    int e1 = (e0 == 0) ? 1 : 0;
#pragma unroll
    for (int e = 0; e < NEXP; ++e) if (e != e0 && p[e] > p[e1]) e1 = e;

    float denom = p[e0] + p[e1] + 1e-8f;
    float w0 = p[e0] / denom, w1 = p[e1] / denom;
    int p0 = atomicAdd(&cnt[e0], 1);
    int p1 = atomicAdd(&cnt[e1], 1);
    eid[2 * t] = e0;  eid[2 * t + 1] = e1;
    wtk[2 * t] = w0;  wtk[2 * t + 1] = w1;
    lpos[2 * t] = p0; lpos[2 * t + 1] = p1;
#pragma unroll
    for (int e = 0; e < NEXP; ++e) atomicAdd(&sP[e], p[e]);
  }
  __syncthreads();
  if (tid < NEXP) atomicAdd(&sumP[tid], sP[tid]);
}

__global__ void prefix_kernel(const int* __restrict__ cnt, int* __restrict__ offsets) {
  if (threadIdx.x == 0 && blockIdx.x == 0) {
    int a = 0;
    for (int e = 0; e < NEXP; ++e) { offsets[e] = a; a += cnt[e]; }
    offsets[NEXP] = a;   // == 2*NTOK
  }
}

__global__ __launch_bounds__(256) void slotfix_kernel(
    const int* __restrict__ eid, const int* __restrict__ lpos,
    const int* __restrict__ offsets, int* __restrict__ slots,
    int* __restrict__ tok_of_slot) {
  int id = blockIdx.x * 256 + threadIdx.x;           // [0, 2N)
  int s = offsets[eid[id]] + lpos[id];
  slots[id] = s;
  tok_of_slot[s] = id >> 1;
}

// ---------------------------------------------------------------------------
// fp32 [R,C] -> bf16 [C,R] tiled transpose+convert (weights -> K-contiguous)
// ---------------------------------------------------------------------------
__global__ __launch_bounds__(256) void transpose_cvt_kernel(
    const float* __restrict__ in, unsigned short* __restrict__ out,
    int R, int C) {
  __shared__ float tile[32][33];
  const int tx = threadIdx.x & 31, ty = threadIdx.x >> 5;  // ty in [0,8)
  const int c0 = blockIdx.x * 32, r0 = blockIdx.y * 32;
#pragma unroll
  for (int i = 0; i < 4; ++i) {
    int r = ty + i * 8;
    tile[r][tx] = in[(size_t)(r0 + r) * C + (c0 + tx)];
  }
  __syncthreads();
#pragma unroll
  for (int i = 0; i < 4; ++i) {
    int c = ty + i * 8;
    out[(size_t)(c0 + c) * R + (r0 + tx)] = f2bf(tile[tx][c]);
  }
}

// ---------------------------------------------------------------------------
// Expert SwiGLU hidden GEMM: h = silu(Xg @ Wg) * (Xg @ Wu)
// Block tile 64(M) x 128(N), K-step 32, double-buffered LDS:
//   A (gathered token rows) via async-to-LDS, Wg/Wu tiles via TDM.
// ---------------------------------------------------------------------------
__global__ __launch_bounds__(256) void moe_hidden_kernel(
    const unsigned short* __restrict__ xb,   // bf16 [N,D]
    const unsigned short* __restrict__ wgT,  // bf16 [F,D]
    const unsigned short* __restrict__ wuT,  // bf16 [F,D]
    unsigned short* __restrict__ h,          // bf16 [N,F] expert-local rows
    const int* __restrict__ tok_of_slot, const int* __restrict__ offsets,
    int expert) {
  __shared__ __align__(16) unsigned short As[2][64][32];
  __shared__ __align__(16) unsigned short Bg[2][128][32];
  __shared__ __align__(16) unsigned short Bu[2][128][32];
  __shared__ int toks[64];

  const int baseS = offsets[expert];
  const int M = offsets[expert + 1] - baseS;
  const int row0 = blockIdx.y * 64;
  if (row0 >= M) return;                      // routed: skip empty tiles
  const int col0 = blockIdx.x * 128;

  const int tid = threadIdx.x;
  if (tid < 64) {
    int r = row0 + tid;
    toks[tid] = (r < M) ? tok_of_slot[baseS + r] : 0;
  }
  __syncthreads();

  const int wave = tid >> 5, lane = tid & 31;
  const int wm = wave >> 1, wn = wave & 1;
  const int lhalf = lane >> 4, l16 = lane & 15;

  v8f cg[4], cu[4];
#pragma unroll
  for (int i = 0; i < 4; ++i)
#pragma unroll
    for (int j = 0; j < 8; ++j) { cg[i][j] = 0.0f; cu[i][j] = 0.0f; }

  const int arow = tid >> 2;                 // 0..63
  const int acol = (tid & 3) * 8;            // 0,8,16,24 (bf16 elems)
  const unsigned short* aptr = xb + (size_t)toks[arow] * DDIM + acol;

  const unsigned tileRows =
      (unsigned)((FDIM - col0) < 128 ? (FDIM - col0) : 128);
#if !HAVE_TDM
  const int brow = tid >> 1;                 // 0..127
  const int bcol = (tid & 1) * 16;           // 0,16
  const int nglob = col0 + brow;
  const bool nvalid = nglob < FDIM;
  const unsigned short* bgp = wgT + (size_t)(nvalid ? nglob : 0) * DDIM + bcol;
  const unsigned short* bup = wuT + (size_t)(nvalid ? nglob : 0) * DDIM + bcol;
  const uint4 zz = make_uint4(0u, 0u, 0u, 0u);
#endif

  auto issue = [&](int kk, int b) {
    const int k0 = kk * 32;
    copy16(aptr + k0, &As[b][arow][acol]);
#if HAVE_TDM
    if (tid < 32) {   // one wave issues the two weight-tile DMAs
      tdm_load_2d(wgT + (size_t)col0 * DDIM + k0, lds_off32(&Bg[b][0][0]),
                  tileRows, DDIM);
      tdm_load_2d(wuT + (size_t)col0 * DDIM + k0, lds_off32(&Bu[b][0][0]),
                  tileRows, DDIM);
    }
#else
    if (nvalid) {
      copy16(bgp + k0, &Bg[b][brow][bcol]);
      copy16(bgp + k0 + 8, &Bg[b][brow][bcol + 8]);
      copy16(bup + k0, &Bu[b][brow][bcol]);
      copy16(bup + k0 + 8, &Bu[b][brow][bcol + 8]);
    } else {
      *(uint4*)&Bg[b][brow][bcol] = zz; *(uint4*)&Bg[b][brow][bcol + 8] = zz;
      *(uint4*)&Bu[b][brow][bcol] = zz; *(uint4*)&Bu[b][brow][bcol + 8] = zz;
    }
#endif
  };

  auto compute = [&](int b) {
    FragU fa;                                // A 16x32 bf16, k = 16*lhalf + s (contiguous)
    fa.u4[0] = *(const uint4*)&As[b][wm * 16 + l16][lhalf * 16];
    fa.u4[1] = *(const uint4*)&As[b][wm * 16 + l16][lhalf * 16 + 8];
#pragma unroll
    for (int tn = 0; tn < 4; ++tn) {
      const int nr = wn * 64 + tn * 16 + l16;
      FragU fb;                              // B 32x16 bf16, k = 16*lhalf + s (contiguous)
      fb.u4[0] = *(const uint4*)&Bg[b][nr][lhalf * 16];
      fb.u4[1] = *(const uint4*)&Bg[b][nr][lhalf * 16 + 8];
      cg[tn] = __builtin_amdgcn_wmma_f32_16x16x32_bf16(
          false, fa.v, false, fb.v, (short)0, cg[tn], false, false);
      fb.u4[0] = *(const uint4*)&Bu[b][nr][lhalf * 16];
      fb.u4[1] = *(const uint4*)&Bu[b][nr][lhalf * 16 + 8];
      cu[tn] = __builtin_amdgcn_wmma_f32_16x16x32_bf16(
          false, fa.v, false, fb.v, (short)0, cu[tn], false, false);
    }
  };

  constexpr int KITER = DDIM / 32;
  issue(0, 0);
  for (int kk = 0; kk < KITER - 1; ++kk) {
    const int b = kk & 1;
    issue(kk + 1, b ^ 1);                    // overlap next-tile DMA
    WAIT_ASYNC(K1_ASY);
#if HAVE_TDM
    if (tid < 32) WAIT_TENSOR(2);
#endif
    __syncthreads();
    compute(b);
    __syncthreads();
  }
  WAIT_ASYNC(0);
#if HAVE_TDM
  if (tid < 32) WAIT_TENSOR(0);
#endif
  __syncthreads();
  compute((KITER - 1) & 1);

#pragma unroll
  for (int tn = 0; tn < 4; ++tn) {
#pragma unroll
    for (int r = 0; r < 8; ++r) {
      const int m = wm * 16 + lhalf * 8 + r;         // C/D VGPR layout
      const int n = col0 + wn * 64 + tn * 16 + l16;
      float g = cg[tn][r], u = cu[tn][r];
      float hv = (g / (1.0f + __expf(-g))) * u;      // silu(g) * u
      if (row0 + m < M && n < FDIM)
        h[(size_t)(row0 + m) * FDIM + n] = f2bf(hv);
    }
  }
}

// ---------------------------------------------------------------------------
// Expert down GEMM: hd[slot,:] = h @ Wd   (contiguous rows per expert)
// ---------------------------------------------------------------------------
__global__ __launch_bounds__(256) void moe_down_kernel(
    const unsigned short* __restrict__ h,    // bf16 [N,F] expert-local rows
    const unsigned short* __restrict__ wdT,  // bf16 [D,F] (K-contig)
    float* __restrict__ hd,                  // [2N, D]
    const int* __restrict__ offsets, int expert) {
  __shared__ __align__(16) unsigned short As[2][64][32];
  __shared__ __align__(16) unsigned short Bd[2][128][32];

  const int baseS = offsets[expert];
  const int M = offsets[expert + 1] - baseS;
  const int row0 = blockIdx.y * 64;
  if (row0 >= M) return;
  const int col0 = blockIdx.x * 128;

  const int tid = threadIdx.x;
  const int wave = tid >> 5, lane = tid & 31;
  const int wm = wave >> 1, wn = wave & 1;
  const int lhalf = lane >> 4, l16 = lane & 15;

  v8f cd[4];
#pragma unroll
  for (int i = 0; i < 4; ++i)
#pragma unroll
    for (int j = 0; j < 8; ++j) cd[i][j] = 0.0f;

  const int arow = tid >> 2, acol = (tid & 3) * 8;
  const unsigned short* aptr = h + (size_t)(row0 + arow) * FDIM + acol;
#if !HAVE_TDM
  const int brow = tid >> 1, bcol = (tid & 1) * 16;
  const unsigned short* bdp = wdT + (size_t)(col0 + brow) * FDIM + bcol;
#endif

  auto issue = [&](int kk, int b) {
    const int k0 = kk * 32;
    copy16(aptr + k0, &As[b][arow][acol]);
#if HAVE_TDM
    if (tid < 32)
      tdm_load_2d(wdT + (size_t)col0 * FDIM + k0, lds_off32(&Bd[b][0][0]),
                  128u, FDIM);
#else
    copy16(bdp + k0, &Bd[b][brow][bcol]);
    copy16(bdp + k0 + 8, &Bd[b][brow][bcol + 8]);
#endif
  };

  auto compute = [&](int b) {
    FragU fa;
    fa.u4[0] = *(const uint4*)&As[b][wm * 16 + l16][lhalf * 16];
    fa.u4[1] = *(const uint4*)&As[b][wm * 16 + l16][lhalf * 16 + 8];
#pragma unroll
    for (int tn = 0; tn < 4; ++tn) {
      const int nr = wn * 64 + tn * 16 + l16;
      FragU fb;
      fb.u4[0] = *(const uint4*)&Bd[b][nr][lhalf * 16];
      fb.u4[1] = *(const uint4*)&Bd[b][nr][lhalf * 16 + 8];
      cd[tn] = __builtin_amdgcn_wmma_f32_16x16x32_bf16(
          false, fa.v, false, fb.v, (short)0, cd[tn], false, false);
    }
  };

  constexpr int KITER = FDIM / 32;           // 86
  issue(0, 0);
  for (int kk = 0; kk < KITER - 1; ++kk) {
    const int b = kk & 1;
    issue(kk + 1, b ^ 1);
    WAIT_ASYNC(K2_ASY);
#if HAVE_TDM
    if (tid < 32) WAIT_TENSOR(1);
#endif
    __syncthreads();
    compute(b);
    __syncthreads();
  }
  WAIT_ASYNC(0);
#if HAVE_TDM
  if (tid < 32) WAIT_TENSOR(0);
#endif
  __syncthreads();
  compute((KITER - 1) & 1);

#pragma unroll
  for (int tn = 0; tn < 4; ++tn) {
#pragma unroll
    for (int r = 0; r < 8; ++r) {
      const int m = wm * 16 + lhalf * 8 + r;
      const int n = col0 + wn * 64 + tn * 16 + l16;
      if (row0 + m < M)
        hd[(size_t)(baseS + row0 + m) * DDIM + n] = cd[tn][r];
    }
  }
}

// ---------------------------------------------------------------------------
// Combine: out[t,d] = w0*hd[slot0,d] + w1*hd[slot1,d]  (deterministic)
// ---------------------------------------------------------------------------
__global__ __launch_bounds__(256) void combine_kernel(
    const float* __restrict__ hd, const int* __restrict__ slots,
    const float* __restrict__ wtk, float* __restrict__ out) {
  const size_t id = (size_t)blockIdx.x * 256 + threadIdx.x;
  const int t = (int)(id >> 10);             // D = 1024
  const int d = (int)(id & 1023);
  float r = wtk[2 * t]     * hd[(size_t)slots[2 * t]     * DDIM + d] +
            wtk[2 * t + 1] * hd[(size_t)slots[2 * t + 1] * DDIM + d];
  out[id] = r;
}

__global__ void aux_kernel(const int* __restrict__ cnt,
                           const float* __restrict__ sumP,
                           float* __restrict__ auxOut) {
  if (threadIdx.x == 0 && blockIdx.x == 0) {
    float a = 0.0f;
    for (int e = 0; e < NEXP; ++e) {
      float frac = (float)cnt[e] / (float)(NTOK * 2);   // one-hot mean
      float load = sumP[e] / (float)NTOK;               // prob mean
      a += frac * load;
    }
    auxOut[0] = 0.01f * (float)NEXP * a;
  }
}

// ---------------------------------------------------------------------------
extern "C" void kernel_launch(void* const* d_in, const int* in_sizes, int n_in,
                              void* d_out, int out_size, void* d_ws,
                              size_t ws_size, hipStream_t stream) {
  (void)in_sizes; (void)n_in; (void)out_size; (void)ws_size;
  const float* x      = (const float*)d_in[0];
  const float* gw     = (const float*)d_in[1];
  const float* w_gate = (const float*)d_in[2];
  const float* w_up   = (const float*)d_in[3];
  const float* w_down = (const float*)d_in[4];
  float* out = (float*)d_out;

  char* p = (char*)d_ws;
  auto alloc = [&](size_t bytes) -> char* {
    char* r = p;
    p += (bytes + 255) & ~(size_t)255;
    return r;
  };
  unsigned short* xb  = (unsigned short*)alloc((size_t)NTOK * DDIM * 2);
  unsigned short* wgT = (unsigned short*)alloc((size_t)FDIM * DDIM * 2);
  unsigned short* wuT = (unsigned short*)alloc((size_t)FDIM * DDIM * 2);
  unsigned short* wdT = (unsigned short*)alloc((size_t)DDIM * FDIM * 2);
  unsigned short* h   = (unsigned short*)alloc((size_t)NTOK * FDIM * 2);
  float* hd   = (float*)alloc((size_t)2 * NTOK * DDIM * 4);
  float* wtk  = (float*)alloc((size_t)NTOK * 2 * 4);
  int*   eid  = (int*)alloc((size_t)NTOK * 2 * 4);
  int*   lpos = (int*)alloc((size_t)NTOK * 2 * 4);
  int*   slots = (int*)alloc((size_t)NTOK * 2 * 4);
  int*   tok_of_slot = (int*)alloc((size_t)NTOK * 2 * 4);
  char*  zr = alloc(256);
  int*   cnt     = (int*)zr;          // 8 ints
  float* sumP    = (float*)(zr + 64); // 8 floats
  int*   offsets = (int*)(zr + 128);  // 9 ints

  (void)hipMemsetAsync(zr, 0, 256, stream);

  gate_kernel<<<NTOK / 8, 256, 0, stream>>>(x, gw, xb, wtk, eid, lpos, cnt, sumP);
  prefix_kernel<<<1, 1, 0, stream>>>(cnt, offsets);
  slotfix_kernel<<<(2 * NTOK) / 256, 256, 0, stream>>>(eid, lpos, offsets, slots,
                                                       tok_of_slot);

  for (int e = 0; e < NEXP; ++e) {
    transpose_cvt_kernel<<<dim3(FDIM / 32, DDIM / 32), 256, 0, stream>>>(
        w_gate + (size_t)e * DDIM * FDIM, wgT, DDIM, FDIM);
    transpose_cvt_kernel<<<dim3(FDIM / 32, DDIM / 32), 256, 0, stream>>>(
        w_up + (size_t)e * DDIM * FDIM, wuT, DDIM, FDIM);
    transpose_cvt_kernel<<<dim3(DDIM / 32, FDIM / 32), 256, 0, stream>>>(
        w_down + (size_t)e * FDIM * DDIM, wdT, FDIM, DDIM);
    moe_hidden_kernel<<<dim3((FDIM + 127) / 128, NTOK / 64), 256, 0, stream>>>(
        xb, wgT, wuT, h, tok_of_slot, offsets, e);
    moe_down_kernel<<<dim3(DDIM / 128, NTOK / 64), 256, 0, stream>>>(
        h, wdT, hd, offsets, e);
  }

  combine_kernel<<<(int)(((size_t)NTOK * DDIM) / 256), 256, 0, stream>>>(
      hd, slots, wtk, out);
  aux_kernel<<<1, 1, 0, stream>>>(cnt, sumP, out + (size_t)NTOK * DDIM);
}
